// DWM_30202210025623
// MI455X (gfx1250) — compile-verified, matrix operand order checked by reading, not run on hardware
//
#include <hip/hip_runtime.h>
#include <hip/hip_bf16.h>

// ---------------- model dims ----------------
#define BB    8
#define SS    96
#define INW   128
#define HH    2
#define MM    32
#define NN    512
#define ST    256
#define OUTW  126
#define CIN   448          // IN + H*M + STATE = 128+64+256
#define NCOL  608          // 256 + 126 + 212 = 594, padded to 38*16
#define KT    14           // 448 / 32
#define NT    38           // 608 / 16
#define EPSF  1e-12f

typedef __attribute__((ext_vector_type(16))) _Float16 v16h;
typedef __attribute__((ext_vector_type(8)))  float    v8f;

// K-offset pattern of the 16-bit 16x32 WMMA fragment (CDNA5 ISA 7.12.2):
// lane-half g (0: lanes 0-15, 1: lanes 16-31), half-index i in v16h.
// For fixed g: i=0..7 -> k = 8*g + 0..7 ; i=8..15 -> k = 16 + 8*g + 0..7
// (two contiguous 8-half chunks -> two ds_load_b128 per lane).
__device__ inline int kpat(int g, int i) {
    int v = i >> 1, p = i & 1;
    int base = (v < 4) ? (2 * v) : (16 + 2 * (v - 4));
    return base + 8 * g + p;
}

__device__ inline float sigm_(float x)  { return 1.f / (1.f + __expf(-x)); }
__device__ inline float sftp_(float x)  { return (x > 20.f) ? x : log1pf(__expf(x)); }
__device__ inline float wsum_(float v)  { for (int o = 16; o > 0; o >>= 1) v += __shfl_xor(v, o, 32); return v; }
__device__ inline float wmax_(float v)  { for (int o = 16; o > 0; o >>= 1) v = fmaxf(v, __shfl_xor(v, o, 32)); return v; }

// ---------------- prep: swizzle fused weights to f16 WMMA B-fragments, init mem ----------------
__global__ void dwm_prep(const float* __restrict__ Ws, const float* __restrict__ Wo,
                         const float* __restrict__ Wu, _Float16* __restrict__ wq,
                         float* __restrict__ mem) {
    int tid = blockIdx.x * blockDim.x + threadIdx.x;
    int nth = gridDim.x * blockDim.x;
    const int WTOT = KT * NT * 32 * 16;
    for (int e = tid; e < WTOT; e += nth) {
        int i = e & 15;
        int l = (e >> 4) & 31;
        int tile = e >> 9;                 // kt*NT + nt
        int nt = tile % NT, kt = tile / NT;
        int n = nt * 16 + (l & 15);
        int k = kt * 32 + kpat(l >> 4, i);
        float w = 0.f;
        if      (n < 256) w = Ws[k * 256 + n];
        else if (n < 382) w = Wo[k * 126 + (n - 256)];
        else if (n < 594) w = Wu[k * 212 + (n - 382)];
        wq[e] = (_Float16)w;
    }
    for (int e = tid; e < BB * NN * MM; e += nth) mem[e] = 0.01f;
}

// ---------------- persistent main kernel: one workgroup, 16 waves (wave32) ----------------
struct Prm { float er[32]; float ad[32]; float kn[32]; float sc[16]; };

__global__ __launch_bounds__(512, 1)
void dwm_main(const float* __restrict__ x,  const float* __restrict__ bs,
              const float* __restrict__ bo, const float* __restrict__ bu,
              const _Float16* __restrict__ wq, float* __restrict__ mem,
              float* __restrict__ out) {
    __shared__ _Float16 combh[16 * CIN];       // 14 KB  (rows 8..15 zero-padded)
    __shared__ float U[BB * NCOL];             // 19 KB  controller GEMM result
    __shared__ float stateL[BB * ST];          //  8 KB
    __shared__ float readL[BB * 64];           //  2 KB
    __shared__ float wtL[BB * HH * NN];        // 32 KB  address weights
    __shared__ float wtkL[BB * HH * NN];       // 32 KB  content scores
    __shared__ Prm  prm[16];                   //  7 KB  per-(b,h) interface params

    const int tid  = threadIdx.x;
    const int lane = tid & 31;                 // wave32 on gfx1250
    const int w    = tid >> 5;                 // wave id 0..15  == b*2 + h
    const int b_   = w >> 1, h_ = w & 1;

    // ---- init carried state ----
    for (int i = tid; i < BB * ST; i += 512) stateL[i] = 1.f;
    for (int i = tid; i < BB * HH * NN; i += 512) wtL[i] = ((i & (NN - 1)) == 0) ? 1.f : 0.f;
    float dyn[16];                             // wt_dyn for (b_,h_), n = lane*16+j
    #pragma unroll
    for (int j = 0; j < 16; j++) dyn[j] = (lane == 0 && j == 0) ? 1.f : 0.f;
    __syncthreads();

    // ---- initial read = wt0 . mem0 ----
    {
        float acc = 0.f;
        const float* mb = mem + (size_t)b_ * NN * MM;
        const float* wrow = wtL + w * NN;
        for (int n = 0; n < NN; n += 4) {
            acc += wrow[n + 0] * mb[(n + 0) * MM + lane];
            acc += wrow[n + 1] * mb[(n + 1) * MM + lane];
            acc += wrow[n + 2] * mb[(n + 2) * MM + lane];
            acc += wrow[n + 3] * mb[(n + 3) * MM + lane];
        }
        readL[b_ * 64 + h_ * 32 + lane] = acc;
    }
    __syncthreads();

    for (int t = 0; t < SS; t++) {
        // ---- Phase A: comb = [x_t | read | state] -> f16, padded to 16 rows ----
        for (int e = tid; e < 16 * CIN; e += 512) {
            int m = e / CIN, k = e - m * CIN;
            float v = 0.f;
            if (m < BB) {
                if      (k < INW)      v = x[((size_t)m * SS + t) * INW + k];
                else if (k < INW + 64) v = readL[m * 64 + (k - INW)];
                else                   v = stateL[m * ST + (k - INW - 64)];
            }
            combh[e] = (_Float16)v;
        }
        __syncthreads();

        // ---- Phase B: U = comb(16x448) @ W(448x608) via v_wmma_f32_16x16x32_f16 ----
        // One accumulator live at a time: low register pressure, no scratch spills.
        {
            const int mrow = lane & 15;
            const int g    = lane >> 4;
            const _Float16* ab = combh + mrow * CIN;    // A row base in LDS
            for (int nt = w; nt < NT; nt += 16) {       // wave-uniform trip count
                v8f acc = {};
                const _Float16* bp = wq + (size_t)nt * 512 + lane * 16;
                #pragma unroll
                for (int kt = 0; kt < KT; kt++) {
                    v16h af;
                    #pragma unroll
                    for (int i = 0; i < 16; i++)
                        af[i] = ab[kt * 32 + kpat(g, i)];
                    if (kt + 1 < KT)   // prefetch next K-tile (global_prefetch_b8)
                        __builtin_prefetch(bp + (size_t)(kt + 1) * NT * 512, 0, 3);
                    v16h bf = *(const v16h*)(bp + (size_t)kt * NT * 512);
                    acc = __builtin_amdgcn_wmma_f32_16x16x32_f16(
                              false, af, false, bf, (short)0, acc, false, false);
                }
                // C/D layout: lanes 0-15 hold rows M=0..7 (VGPR r -> M=r); need only 8 rows
                if (lane < 16) {
                    #pragma unroll
                    for (int r = 0; r < 8; r++)
                        U[r * NCOL + nt * 16 + lane] = acc[r];
                }
            }
        }
        __syncthreads();

        // ---- Phase C: state/out + interface params ----
        for (int i = tid; i < BB * ST; i += 512) {
            int b = i >> 8, q = i & 255;
            stateL[i] = sigm_(U[b * NCOL + q] + bs[q]);
        }
        for (int i = tid; i < BB * OUTW; i += 512) {
            int b = i / OUTW, q = i - b * OUTW;
            out[((size_t)b * SS + t) * OUTW + q] = U[b * NCOL + 256 + q] + bo[q];
        }
        {
            const int cb = 382 + h_ * 106;     // column base in U
            const int ub = h_ * 106;           // bias base in b_u
            const float* Ub = U + b_ * NCOL;
            float er = sigm_(Ub[cb + 8  + lane] + bu[ub + 8  + lane]);
            float ad =       Ub[cb + 40 + lane] + bu[ub + 40 + lane];   // add: raw
            float kv = tanhf(Ub[cb + 72 + lane] + bu[ub + 72 + lane]);
            float kn = kv / (sqrtf(wsum_(kv * kv)) + EPSF);
            prm[w].er[lane] = er; prm[w].ad[lane] = ad; prm[w].kn[lane] = kn;
            // scalar params (computed redundantly across the wave)
            float s0 = sftp_(Ub[cb + 0] + bu[ub + 0]);
            float s1 = sftp_(Ub[cb + 1] + bu[ub + 1]);
            float s2 = sftp_(Ub[cb + 2] + bu[ub + 2]);
            float sm = fmaxf(s0, fmaxf(s1, s2));
            float e0 = __expf(s0 - sm), e1 = __expf(s1 - sm), e2 = __expf(s2 - sm);
            float es = e0 + e1 + e2;
            float jd = sigm_(Ub[cb + 3] + bu[ub + 3]);
            float q0 = Ub[cb + 4] + bu[ub + 4];
            float q1 = Ub[cb + 5] + bu[ub + 5];
            float q2 = Ub[cb + 6] + bu[ub + 6];
            float qm = fmaxf(q0, fmaxf(q1, q2));
            float f0 = __expf(q0 - qm), f1 = __expf(q1 - qm), f2 = __expf(q2 - qm);
            float fs = f0 + f1 + f2;
            float gamma = 1.f + sftp_(Ub[cb + 7]   + bu[ub + 7]);
            float beta  =       sftp_(Ub[cb + 104] + bu[ub + 104]);
            float gg    = sigm_(Ub[cb + 105] + bu[ub + 105]);
            if (lane == 0) {
                float* sc = prm[w].sc;
                sc[0] = e0 / es; sc[1] = e1 / es; sc[2] = e2 / es;
                sc[3] = jd;      sc[4] = f0 / fs; sc[5] = f1 / fs; sc[6] = f2 / fs;
                sc[7] = gamma;   sc[8] = beta;    sc[9] = gg;
            }
        }
        __syncthreads();

        // ---- Phase D: mem erase/add, column norms, content scores wt_k ----
        for (int cc = tid; cc < BB * NN; cc += 512) {
            int b = cc >> 9, n = cc & (NN - 1);
            float* pm = mem + ((size_t)(b * NN + n)) * MM;     // mem[b][n][*], contiguous
            float col[32];
            #pragma unroll
            for (int m4 = 0; m4 < 8; m4++) {
                float4 v4 = *(const float4*)(pm + m4 * 4);
                col[m4*4+0] = v4.x; col[m4*4+1] = v4.y; col[m4*4+2] = v4.z; col[m4*4+3] = v4.w;
            }
            float w0 = wtL[(b * 2 + 0) * NN + n];
            float w1 = wtL[(b * 2 + 1) * NN + n];
            const Prm& p0 = prm[b * 2 + 0];
            const Prm& p1 = prm[b * 2 + 1];
            float ss = 0.f, d0 = 0.f, d1 = 0.f;
            #pragma unroll
            for (int m = 0; m < 32; m++) {
                float f = (1.f - p0.er[m] * w0) * (1.f - p1.er[m] * w1);
                float v = col[m] * f + p0.ad[m] * w0 + p1.ad[m] * w1;
                col[m] = v; ss += v * v; d0 += p0.kn[m] * v; d1 += p1.kn[m] * v;
            }
            #pragma unroll
            for (int m4 = 0; m4 < 8; m4++) {
                float4 v4 = { col[m4*4+0], col[m4*4+1], col[m4*4+2], col[m4*4+3] };
                *(float4*)(pm + m4 * 4) = v4;
            }
            float inv = 1.f / (sqrtf(ss) + EPSF);
            wtkL[(b * 2 + 0) * NN + n] = d0 * inv;
            wtkL[(b * 2 + 1) * NN + n] = d1 * inv;
        }
        __syncthreads();

        // ---- Phase E: softmax/gate/shift/sharpen/jump (one wave per (b,h)) ----
        {
            const float* sc = prm[w].sc;
            const float s0 = sc[0], s1 = sc[1], s2 = sc[2], jd = sc[3];
            const float j0 = sc[4], j1 = sc[5], j2 = sc[6];
            const float gamma = sc[7], beta = sc[8], gg = sc[9];
            float* wrow = wtL  + w * NN;
            float* krow = wtkL + w * NN;
            float v[16], c[16];
            float mx = -1e30f;
            #pragma unroll
            for (int j = 0; j < 16; j++) { v[j] = beta * krow[lane * 16 + j]; mx = fmaxf(mx, v[j]); }
            mx = wmax_(mx);
            float sum = 0.f;
            #pragma unroll
            for (int j = 0; j < 16; j++) { v[j] = __expf(v[j] - mx); sum += v[j]; }
            sum = wsum_(sum);
            float isum = 1.f / sum;
            #pragma unroll
            for (int j = 0; j < 16; j++)
                c[j] = gg * (v[j] * isum) + (1.f - gg) * wrow[lane * 16 + j];
            // circular conv: conv[n] = s0*wc[n+1] + s1*wc[n] + s2*wc[n-1]
            float cl = __shfl(c[15], (lane + 31) & 31, 32);   // n-1 across lane boundary
            float cr = __shfl(c[0],  (lane + 1)  & 31, 32);   // n+1 across lane boundary
            float wn[16], tot = 0.f;
            #pragma unroll
            for (int j = 0; j < 16; j++) {
                float left  = (j > 0)  ? c[j - 1] : cl;
                float right = (j < 15) ? c[j + 1] : cr;
                float conv  = s0 * right + s1 * c[j] + s2 * left;
                float sh    = __expf(gamma * __logf(conv + EPSF));
                wn[j] = sh; tot += sh;
            }
            tot = wsum_(tot);
            float it = 1.f / tot;
            #pragma unroll
            for (int j = 0; j < 16; j++) {
                float wtn  = wn[j] * it;
                float od   = dyn[j];
                dyn[j]     = (1.f - jd) * od + jd * wtn;
                float a0   = (lane == 0 && j == 0) ? 1.f : 0.f;
                wrow[lane * 16 + j] = j0 * wtn + j1 * a0 + j2 * od;   // uses OLD wt_dyn
            }
        }
        __syncthreads();

        // ---- Phase F: read = wt_f . mem (lane = m, coalesced over m) ----
        {
            float acc = 0.f;
            const float* mb = mem + (size_t)b_ * NN * MM;
            const float* wrow = wtL + w * NN;
            for (int n = 0; n < NN; n += 4) {
                acc += wrow[n + 0] * mb[(n + 0) * MM + lane];
                acc += wrow[n + 1] * mb[(n + 1) * MM + lane];
                acc += wrow[n + 2] * mb[(n + 2) * MM + lane];
                acc += wrow[n + 3] * mb[(n + 3) * MM + lane];
            }
            readL[b_ * 64 + h_ * 32 + lane] = acc;
        }
        __syncthreads();
    }
}

// ---------------- launch ----------------
extern "C" void kernel_launch(void* const* d_in, const int* in_sizes, int n_in,
                              void* d_out, int out_size, void* d_ws, size_t ws_size,
                              hipStream_t stream) {
    const float* x  = (const float*)d_in[0];
    const float* Ws = (const float*)d_in[1];
    const float* bs = (const float*)d_in[2];
    const float* Wo = (const float*)d_in[3];
    const float* bo = (const float*)d_in[4];
    const float* Wu = (const float*)d_in[5];
    const float* bu = (const float*)d_in[6];

    _Float16* wq = (_Float16*)d_ws;
    size_t wqBytes = (size_t)KT * NT * 32 * 16 * sizeof(_Float16);   // 544768 B
    size_t memOff  = (wqBytes + 255) & ~(size_t)255;
    float* mem = (float*)((char*)d_ws + memOff);                     // + 512 KB

    dwm_prep<<<512, 256, 0, stream>>>(Ws, Wo, Wu, wq, mem);
    dwm_main<<<1, 512, 0, stream>>>(x, bs, bo, bu, wq, mem, (float*)d_out);
}
